// GIN_28278064677524
// MI455X (gfx1250) — matrix operand order repaired and checked by reading, not compiled, not run on hardware
//
#include <hip/hip_runtime.h>

// ---------------- problem constants (match reference) ----------------
#define NNODES 50000
#define NEDGES 800000
#define FIN    256
#define FH     512
#define FOUT   128
#define NGRAPH 500
#define BN_INV 0.9999950000374996f   // 1/sqrt(1+1e-5)

// ---------------- bf16 helpers ----------------
typedef __attribute__((ext_vector_type(16))) __bf16 v16bf;
typedef __attribute__((ext_vector_type(8)))  float  v8f;

static __device__ __forceinline__ unsigned short f2bf(float f) {
  union { float f; unsigned int u; } c; c.f = f;
  unsigned int r = c.u + 0x7FFFu + ((c.u >> 16) & 1u);   // round-to-nearest-even
  return (unsigned short)(r >> 16);
}
static __device__ __forceinline__ float bf2f(unsigned short h) {
  union { unsigned int u; float f; } c; c.u = ((unsigned int)h) << 16;
  return c.f;
}
static __device__ __forceinline__ float bflo(unsigned int p) {
  return bf2f((unsigned short)(p & 0xffffu));
}
static __device__ __forceinline__ float bfhi(unsigned int p) {
  return bf2f((unsigned short)(p >> 16));
}
static __device__ __forceinline__ unsigned int packbf(float a, float b) {
  return (unsigned int)f2bf(a) | ((unsigned int)f2bf(b) << 16);
}

// ---- CDNA5 async global->LDS copy (ASYNCcnt path), 32B per lane ----
// INST_OFFSET is added to BOTH the LDS and global addresses (ISA 15.18.3),
// so one base pair covers both 16B halves.
static __device__ __forceinline__ void async_copy32(unsigned lds_addr, const void* gptr) {
  unsigned long long ga = (unsigned long long)(size_t)gptr;
  asm volatile("global_load_async_to_lds_b128 %0, %1, off"
               :: "v"(lds_addr), "v"(ga) : "memory");
  asm volatile("global_load_async_to_lds_b128 %0, %1, off offset:16"
               :: "v"(lds_addr), "v"(ga) : "memory");
}
static __device__ __forceinline__ void wait_async0() {
  asm volatile("s_wait_asynccnt 0x0" ::: "memory");
}

// ---------------- small utility kernels ----------------
__global__ void k_zero_i32(int* p, int n) {
  int i = blockIdx.x * blockDim.x + threadIdx.x;
  if (i < n) p[i] = 0;
}

__global__ void k_count_deg(const int* __restrict__ dst, int* __restrict__ deg, int e) {
  int i = blockIdx.x * blockDim.x + threadIdx.x;
  if (i < e) atomicAdd(&deg[dst[i]], 1);
}

// single-block exclusive scan over deg -> rowptr (size n+1) and cursor copy
__global__ void k_scan_rowptr(const int* __restrict__ deg, int* __restrict__ rowptr,
                              int* __restrict__ cursor, int n) {
  __shared__ int sh[1024];
  __shared__ int s_carry;
  int tid = threadIdx.x;
  if (tid == 0) s_carry = 0;
  __syncthreads();
  for (int base = 0; base < n; base += 1024) {
    int c0 = s_carry;
    int i = base + tid;
    int v = (i < n) ? deg[i] : 0;
    sh[tid] = v;
    __syncthreads();
    #pragma unroll
    for (int off = 1; off < 1024; off <<= 1) {
      int t = (tid >= off) ? sh[tid - off] : 0;
      __syncthreads();
      sh[tid] += t;
      __syncthreads();
    }
    int excl = c0 + sh[tid] - v;
    if (i < n) { rowptr[i] = excl; cursor[i] = excl; }
    int tot = sh[1023];
    __syncthreads();
    if (tid == 0) s_carry = c0 + tot;
    __syncthreads();
  }
  if (tid == 0) rowptr[n] = s_carry;
}

__global__ void k_scatter_edges(const int* __restrict__ src, const int* __restrict__ dst,
                                int* __restrict__ cursor, int* __restrict__ csr_src, int e) {
  int i = blockIdx.x * blockDim.x + threadIdx.x;
  if (i < e) {
    int p = atomicAdd(&cursor[dst[i]], 1);
    csr_src[p] = src[i];
  }
}

// fp32 [K x Nc] row-major  ->  bf16 transposed [Nc x K]
__global__ void k_cvt_transpose_bf16(const float* __restrict__ in,
                                     unsigned short* __restrict__ outT, int K, int Nc) {
  int idx = blockIdx.x * blockDim.x + threadIdx.x;
  if (idx >= K * Nc) return;
  int n = idx / K, k = idx % K;
  outT[idx] = f2bf(in[k * Nc + n]);
}

// ---------------- aggregation: agg[i] = (1+eps)*x[i] + sum_{j in N(i)} x[j] ----------------
// layer-1 variant: fp32 input, C = FIN = 256 (one column per thread)
__global__ __launch_bounds__(256)
void k_aggregate_f32(const float* __restrict__ x,
                     const int* __restrict__ rowptr, const int* __restrict__ csr_src,
                     const float* __restrict__ eps_ptr, unsigned short* __restrict__ agg) {
  __shared__ int nb[256];
  int node = blockIdx.x;
  float epw = 1.0f + eps_ptr[0];
  int lo = rowptr[node], hi = rowptr[node + 1];
  int c = threadIdx.x;
  float a = epw * x[(size_t)node * FIN + c];
  for (int s = lo; s < hi; s += 256) {
    int cnt = hi - s; if (cnt > 256) cnt = 256;
    __syncthreads();
    if ((int)threadIdx.x < cnt) nb[threadIdx.x] = csr_src[s + threadIdx.x];
    __syncthreads();
    for (int k = 0; k < cnt; ++k) a += x[(size_t)nb[k] * FIN + c];
  }
  agg[(size_t)node * FIN + c] = f2bf(a);
}

// later-layer variant: bf16 input, C = FH = 512 (one packed u32 column-pair per thread)
__global__ __launch_bounds__(256)
void k_aggregate_bf16(const unsigned short* __restrict__ x,
                      const int* __restrict__ rowptr, const int* __restrict__ csr_src,
                      const float* __restrict__ eps_ptr, unsigned short* __restrict__ agg) {
  __shared__ int nb[256];
  int node = blockIdx.x;
  float epw = 1.0f + eps_ptr[0];
  int lo = rowptr[node], hi = rowptr[node + 1];
  const unsigned int* xu = (const unsigned int*)x;
  size_t rowu = (size_t)node * (FH / 2) + threadIdx.x;
  unsigned int own = xu[rowu];
  float a0 = epw * bflo(own);
  float a1 = epw * bfhi(own);
  for (int s = lo; s < hi; s += 256) {
    int cnt = hi - s; if (cnt > 256) cnt = 256;
    __syncthreads();
    if ((int)threadIdx.x < cnt) nb[threadIdx.x] = csr_src[s + threadIdx.x];
    __syncthreads();
    for (int k = 0; k < cnt; ++k) {
      unsigned int p = xu[(size_t)nb[k] * (FH / 2) + threadIdx.x];
      a0 += bflo(p);
      a1 += bfhi(p);
    }
  }
  ((unsigned int*)agg)[rowu] = packbf(a0, a1);
}

// ---------------- WMMA GEMM: out = post( A[MxK] @ W[KxN] + bias ) ----------------
// A: bf16 row-major (lda).  WT: bf16, W transposed [Nc x K].
// mode 1: relu ; mode 2: relu then *scale*BN_INV + shift
// Block tile 128x128, 8 waves, wave tile 32x64 (2x4 WMMA accs),
// double-buffered LDS fed by async global->LDS DMA (ASYNCcnt).
#define BM 128
#define BN 128
#define BK 32
#define LSTR 40   // LDS row stride in bf16 elements (80B: 16B-aligned, conflict-free)

__global__ __launch_bounds__(256)
void k_gemm_bf16_wmma(const unsigned short* __restrict__ A, int lda,
                      const unsigned short* __restrict__ WT,
                      const float* __restrict__ bias,
                      const float* __restrict__ scale, const float* __restrict__ shift,
                      unsigned short* __restrict__ out, int ldo,
                      int M, int K, int mode) {
  __shared__ unsigned short ldsA[2][BM * LSTR];
  __shared__ unsigned short ldsB[2][BN * LSTR];

  int tid  = threadIdx.x;
  int wave = tid >> 5;
  int lane = tid & 31;
  int tileM = blockIdx.x * BM;
  int tileN = blockIdx.y * BN;

  int wm = (wave >> 1) * 32;   // wave M offset within tile (0..96)
  int wn = (wave & 1) * 64;    // wave N offset within tile (0/64)

  // global tile loads: 256 threads, each thread owns one half-row (32B)
  int grow = tid >> 1;            // 0..127
  int gcol = (tid & 1) * 16;      // element col 0 / 16
  int arow = tileM + grow; if (arow >= M) arow = M - 1;   // clamp (stores guarded)
  const unsigned short* aptr = A  + (size_t)arow * lda + gcol;
  const unsigned short* bptr = WT + (size_t)(tileN + grow) * K + gcol;
  int lofs = grow * LSTR + gcol;
  unsigned la[2], lb[2];
  la[0] = (unsigned)(size_t)&ldsA[0][lofs];
  la[1] = (unsigned)(size_t)&ldsA[1][lofs];
  lb[0] = (unsigned)(size_t)&ldsB[0][lofs];
  lb[1] = (unsigned)(size_t)&ldsB[1][lofs];

  int khalf = lane >> 4;
  int lm    = lane & 15;
  int aOff0 = (wm + lm) * LSTR;
  int aOff1 = (wm + 16 + lm) * LSTR;
  int bOff[4];
  #pragma unroll
  for (int t = 0; t < 4; ++t) bOff[t] = (wn + 16 * t + lm) * LSTR;

  v8f acc[2][4] = {};

  int nIter = K / BK;
  // prologue: DMA tile 0 into buffer 0
  async_copy32(la[0], aptr);
  async_copy32(lb[0], bptr);

  int cur = 0;
  for (int it = 0; it < nIter; ++it) {
    wait_async0();        // this wave's pending tile copies landed in LDS
    __syncthreads();      // all waves' copies landed; all done reading buf cur^1
    if (it + 1 < nIter) { // DMA next tile into the other buffer, overlapped with WMMAs
      int k0 = (it + 1) * BK;
      async_copy32(la[cur ^ 1], aptr + k0);
      async_copy32(lb[cur ^ 1], bptr + k0);
    }
    // fragment reads from current LDS buffer
    union Frag { v16bf v; unsigned int u[8]; };
    Frag af[2], bfr[4];
    #pragma unroll
    for (int v = 0; v < 8; ++v) {
      // A frag (ISA 16-bit A 16x32 layout): k = (v&3)*2 + (v>>2)*16 + 8*khalf (+e)
      int ka = ((v & 3) * 2) + ((v >> 2) * 16) + 8 * khalf;
      af[0].u[v] = *(const unsigned int*)&ldsA[cur][aOff0 + ka];
      af[1].u[v] = *(const unsigned int*)&ldsA[cur][aOff1 + ka];
      // B frag: same k order as A frag: k = (v&3)*2 + (v>>2)*16 + 8*khalf (+e)
      int kb = ((v & 3) * 2) + ((v >> 2) * 16) + 8 * khalf;
      #pragma unroll
      for (int t = 0; t < 4; ++t)
        bfr[t].u[v] = *(const unsigned int*)&ldsB[cur][bOff[t] + kb];
    }
    #pragma unroll
    for (int i = 0; i < 2; ++i)
      #pragma unroll
      for (int t = 0; t < 4; ++t)
        acc[i][t] = __builtin_amdgcn_wmma_f32_16x16x32_bf16(
            false, af[i].v, false, bfr[t].v, (short)0, acc[i][t], false, false);
    cur ^= 1;
  }

  // epilogue: C/D layout — VGPR r holds M=r (lanes 0-15) / M=r+8 (lanes 16-31), N=lane&15
  float bs[4], sc[4], sf[4];
  #pragma unroll
  for (int t = 0; t < 4; ++t) {
    int n = tileN + wn + 16 * t + lm;
    bs[t] = bias[n];
    sc[t] = (mode == 2) ? scale[n] * BN_INV : 1.0f;
    sf[t] = (mode == 2) ? shift[n] : 0.0f;
  }
  #pragma unroll
  for (int i = 0; i < 2; ++i) {
    #pragma unroll
    for (int r = 0; r < 8; ++r) {
      int m_abs = tileM + wm + 16 * i + r + 8 * khalf;
      if (m_abs < M) {
        #pragma unroll
        for (int t = 0; t < 4; ++t) {
          int n = tileN + wn + 16 * t + lm;
          float v = fmaxf(acc[i][t][r] + bs[t], 0.0f);
          v = v * sc[t] + sf[t];
          out[(size_t)m_abs * ldo + n] = f2bf(v);
        }
      }
    }
  }
}

// ---------------- mean pool over sorted batch ----------------
static __device__ __forceinline__ int dlower_bound(const int* a, int n, int key) {
  int lo = 0, hi = n;
  while (lo < hi) { int mid = (lo + hi) >> 1; if (a[mid] < key) lo = mid + 1; else hi = mid; }
  return lo;
}

__global__ __launch_bounds__(256)
void k_pool(const unsigned short* __restrict__ x, const int* __restrict__ batch,
            float* __restrict__ xm) {
  __shared__ int s_lo, s_hi;
  int g = blockIdx.x;
  if (threadIdx.x == 0) {
    s_lo = dlower_bound(batch, NNODES, g);
    s_hi = dlower_bound(batch, NNODES, g + 1);
  }
  __syncthreads();
  int lo = s_lo, hi = s_hi;
  int cnt = hi - lo; if (cnt < 1) cnt = 1;
  float inv = 1.0f / (float)cnt;
  const unsigned int* xu = (const unsigned int*)x;
  int cu = threadIdx.x;                 // u32 column pair, 256 covers FH=512
  float a0 = 0.0f, a1 = 0.0f;
  for (int i = lo; i < hi; ++i) {
    unsigned int p = xu[(size_t)i * (FH / 2) + cu];
    a0 += bflo(p);
    a1 += bfhi(p);
  }
  xm[g * FH + 2 * cu]     = a0 * inv;
  xm[g * FH + 2 * cu + 1] = a1 * inv;
}

// ---------------- fp32 head linear ----------------
__global__ void k_linear(const float* __restrict__ X, const float* __restrict__ W,
                         const float* __restrict__ b, float* __restrict__ Y,
                         int M, int K, int Nc, int relu) {
  int idx = blockIdx.x * blockDim.x + threadIdx.x;
  if (idx >= M * Nc) return;
  int m = idx / Nc, n = idx % Nc;
  float acc = b[n];
  for (int k = 0; k < K; ++k) acc += X[m * K + k] * W[k * Nc + n];
  Y[idx] = relu ? fmaxf(acc, 0.0f) : acc;
}

// ---------------- log_softmax over 128-wide rows ----------------
__global__ void k_logsoftmax(const float* __restrict__ logits, float* __restrict__ out) {
  __shared__ float red[FOUT];
  int g = blockIdx.x, t = threadIdx.x;
  float v = logits[g * FOUT + t];
  red[t] = v; __syncthreads();
  for (int off = FOUT / 2; off > 0; off >>= 1) {
    if (t < off) red[t] = fmaxf(red[t], red[t + off]);
    __syncthreads();
  }
  float mx = red[0]; __syncthreads();
  float e = __expf(v - mx);
  red[t] = e; __syncthreads();
  for (int off = FOUT / 2; off > 0; off >>= 1) {
    if (t < off) red[t] += red[t + off];
    __syncthreads();
  }
  out[g * FOUT + t] = v - mx - logf(red[0]);
}

// ---------------- host-side orchestration ----------------
extern "C" void kernel_launch(void* const* d_in, const int* in_sizes, int n_in,
                              void* d_out, int out_size, void* d_ws, size_t ws_size,
                              hipStream_t stream) {
  (void)in_sizes; (void)n_in; (void)out_size; (void)ws_size;

  const float* x_in  = (const float*)d_in[0];
  const int*   ei    = (const int*)d_in[1];
  const int*   srcE  = ei;
  const int*   dstE  = ei + NEDGES;
  const int*   batch = (const int*)d_in[2];
  const float* w1a = (const float*)d_in[3];  const float* b1a = (const float*)d_in[4];
  const float* w1b = (const float*)d_in[5];  const float* b1b = (const float*)d_in[6];
  const float* g1  = (const float*)d_in[7];  const float* be1 = (const float*)d_in[8];
  const float* eps1= (const float*)d_in[9];
  const float* wa  = (const float*)d_in[10]; const float* ba  = (const float*)d_in[11];
  const float* wb  = (const float*)d_in[12]; const float* bb  = (const float*)d_in[13];
  const float* gg  = (const float*)d_in[14]; const float* be  = (const float*)d_in[15];
  const float* eps = (const float*)d_in[16];
  const float* l1w = (const float*)d_in[17]; const float* l1b = (const float*)d_in[18];
  const float* l2w = (const float*)d_in[19]; const float* l2b = (const float*)d_in[20];

  // ---- workspace layout (256B aligned) ----
  char* base = (char*)d_ws;
  size_t off = 0;
  auto take = [&](size_t bytes) -> char* {
    char* p = base + off;
    off += (bytes + 255) & ~(size_t)255;
    return p;
  };
  unsigned short* w1aT = (unsigned short*)take((size_t)FH * FIN * 2);   // [512 x 256]
  unsigned short* w1bT = (unsigned short*)take((size_t)FH * FH * 2);    // [512 x 512]
  unsigned short* waT  = (unsigned short*)take((size_t)3 * FH * FH * 2);
  unsigned short* wbT  = (unsigned short*)take((size_t)3 * FH * FH * 2);
  int* rowptr  = (int*)take((size_t)(NNODES + 1) * 4);
  int* deg     = (int*)take((size_t)NNODES * 4);
  int* cursor  = (int*)take((size_t)NNODES * 4);
  int* csr_src = (int*)take((size_t)NEDGES * 4);
  unsigned short* XBUF = (unsigned short*)take((size_t)NNODES * FH * 2);
  unsigned short* AGG  = (unsigned short*)take((size_t)NNODES * FH * 2);
  unsigned short* Ybuf = (unsigned short*)take((size_t)NNODES * FH * 2);
  float* xm     = (float*)take((size_t)NGRAPH * FH * 4);
  float* hh     = (float*)take((size_t)NGRAPH * FH * 4);
  float* logits = (float*)take((size_t)NGRAPH * FOUT * 4);

  // ---- weight convert + transpose (bf16) ----
  k_cvt_transpose_bf16<<<(FIN * FH + 255) / 256, 256, 0, stream>>>(w1a, w1aT, FIN, FH);
  k_cvt_transpose_bf16<<<(FH * FH + 255) / 256, 256, 0, stream>>>(w1b, w1bT, FH, FH);
  for (int i = 0; i < 3; ++i) {
    k_cvt_transpose_bf16<<<(FH * FH + 255) / 256, 256, 0, stream>>>(
        wa + (size_t)i * FH * FH, waT + (size_t)i * FH * FH, FH, FH);
    k_cvt_transpose_bf16<<<(FH * FH + 255) / 256, 256, 0, stream>>>(
        wb + (size_t)i * FH * FH, wbT + (size_t)i * FH * FH, FH, FH);
  }

  // ---- CSR by destination ----
  k_zero_i32<<<(NNODES + 255) / 256, 256, 0, stream>>>(deg, NNODES);
  k_count_deg<<<(NEDGES + 255) / 256, 256, 0, stream>>>(dstE, deg, NEDGES);
  k_scan_rowptr<<<1, 1024, 0, stream>>>(deg, rowptr, cursor, NNODES);
  k_scatter_edges<<<(NEDGES + 255) / 256, 256, 0, stream>>>(srcE, dstE, cursor, csr_src, NEDGES);

  dim3 gemmGrid((NNODES + BM - 1) / BM, FH / BN);

  // ---- layer 1 (K = 256) ----
  k_aggregate_f32<<<NNODES, 256, 0, stream>>>(x_in, rowptr, csr_src, eps1, AGG);
  k_gemm_bf16_wmma<<<gemmGrid, 256, 0, stream>>>(AGG, FIN, w1aT, b1a, nullptr, nullptr,
                                                 Ybuf, FH, NNODES, FIN, 1);
  k_gemm_bf16_wmma<<<gemmGrid, 256, 0, stream>>>(Ybuf, FH, w1bT, b1b, g1, be1,
                                                 XBUF, FH, NNODES, FH, 2);

  // ---- layers 2..4 (K = 512) ----
  for (int i = 0; i < 3; ++i) {
    k_aggregate_bf16<<<NNODES, 256, 0, stream>>>(XBUF, rowptr, csr_src, eps + i, AGG);
    k_gemm_bf16_wmma<<<gemmGrid, 256, 0, stream>>>(AGG, FH, waT + (size_t)i * FH * FH,
                                                   ba + (size_t)i * FH, nullptr, nullptr,
                                                   Ybuf, FH, NNODES, FH, 1);
    k_gemm_bf16_wmma<<<gemmGrid, 256, 0, stream>>>(Ybuf, FH, wbT + (size_t)i * FH * FH,
                                                   bb + (size_t)i * FH,
                                                   gg + (size_t)i * FH, be + (size_t)i * FH,
                                                   XBUF, FH, NNODES, FH, 2);
  }

  // ---- pool + head ----
  k_pool<<<NGRAPH, 256, 0, stream>>>(XBUF, batch, xm);
  k_linear<<<(NGRAPH * FH + 255) / 256, 256, 0, stream>>>(xm, l1w, l1b, hh, NGRAPH, FH, FH, 1);
  k_linear<<<(NGRAPH * FOUT + 255) / 256, 256, 0, stream>>>(hh, l2w, l2b, logits, NGRAPH, FH, FOUT, 0);
  k_logsoftmax<<<NGRAPH, FOUT, 0, stream>>>(logits, (float*)d_out);
}
